// UnifiedBlock_38611755991500
// MI455X (gfx1250) — compile-verified, matrix operand order checked by reading, not run on hardware
//
#include <hip/hip_runtime.h>
#include <hip/hip_bf16.h>

// ---------------------------------------------------------------------------
// Types for CDNA5 WMMA / TDM
// ---------------------------------------------------------------------------
typedef __attribute__((ext_vector_type(16))) _Float16     v16h;
typedef __attribute__((ext_vector_type(8)))  float        v8f;
typedef __attribute__((ext_vector_type(4)))  unsigned int u32x4;
typedef __attribute__((ext_vector_type(8)))  int          i32x8;
typedef __attribute__((ext_vector_type(4)))  int          i32x4;

#define BDIM 256

// Problem constants (match reference)
#define B_  4
#define S_  2048
#define H_  1024
#define NH_ 16
#define HD_ 64
#define I_  2764
#define IP_ 2816              // I padded to multiple of 128 (and 32)
#define K_  4
#define MS_ (B_ * S_)         // 8192 rows (multiple of 128)
#define H3_ (3 * H_)          // 3072
#define H2_ (2 * H_)          // 2048

// ---------------------------------------------------------------------------
// Tensor Data Mover: 2D tile load Global -> LDS.
// Builds D# group0/group1 per CDNA5 ISA §8 (CDNA5 ISA) and issues
// tensor_load_to_lds. data_size = 2 bytes (f16). LDS row padding:
// pad_interval = 16 DWORDs (one 32-half row), pad_amount = 8 DWORDs (16
// halfs) -> LDS rows land 96 B apart (LDT = 48 halfs).
// ---------------------------------------------------------------------------
__device__ __forceinline__ void tdm_load_tile_f16(
    const _Float16* gsrc, unsigned lds_byte_addr,
    int tile_w, int tile_h, int tensor_w, int tensor_h, int row_stride) {
  unsigned long long ga = (unsigned long long)(const void*)gsrc;

  u32x4 g0;
  g0[0] = 1u;                                        // count=1, user descriptor
  g0[1] = lds_byte_addr;                             // lds_addr [63:32]
  g0[2] = (unsigned)(ga & 0xffffffffull);            // global_addr[31:0]
  g0[3] = (unsigned)((ga >> 32) & 0x01ffffffull)     // global_addr[56:32]
        | (2u << 30);                                // type = 2 ("image")

  i32x8 g1;
  unsigned w0 = (1u << 16)                           // data_size = 1 -> 2 bytes
              | (1u << 20)                           // pad_enable
              | (3u << 22)                           // pad_interval: 16 DWORDs
              | (7u << 25);                          // pad_amount : 8 DWORDs
  g1[0] = (int)w0;
  g1[1] = (int)(((unsigned)tensor_w & 0xffffu) << 16);          // tensor_dim0[15:0]
  g1[2] = (int)((((unsigned)tensor_w >> 16) & 0xffffu)          // tensor_dim0[31:16]
        |        (((unsigned)tensor_h & 0xffffu) << 16));       // tensor_dim1[15:0]
  g1[3] = (int)((((unsigned)tensor_h >> 16) & 0xffffu)          // tensor_dim1[31:16]
        |        (((unsigned)tile_w & 0xffffu) << 16));         // tile_dim0
  g1[4] = (int)((unsigned)tile_h & 0xffffu);                    // tile_dim1 (dim2=0)
  g1[5] = (int)(unsigned)row_stride;                            // dim0_stride[31:0]
  g1[6] = 0;                                                    // stride hi / dim1_stride
  g1[7] = 0;

  i32x4 z4 = {0, 0, 0, 0};
#if __has_include(<hip/amd_detail/amd_gfx1250_TDM.h>)
  i32x8 z8 = {0, 0, 0, 0, 0, 0, 0, 0};
  __builtin_amdgcn_tensor_load_to_lds(g0, g1, z4, z4, z8, 0);   // 6-arg toolchain
#else
  __builtin_amdgcn_tensor_load_to_lds(g0, g1, z4, z4, 0);       // 5-arg (ROCm 7.2)
#endif
}

// ---------------------------------------------------------------------------
// f32 -> f16 conversion
// ---------------------------------------------------------------------------
__global__ __launch_bounds__(BDIM) void cvt_f16_kernel(const float* __restrict__ src,
                                                       _Float16* __restrict__ dst,
                                                       size_t n) {
  size_t i = (size_t)blockIdx.x * BDIM + threadIdx.x;
  size_t stride = (size_t)gridDim.x * BDIM;
  for (; i < n; i += stride) dst[i] = (_Float16)src[i];
}

// f32 [src_rows, src_cols] -> f16 [dst_rows, dst_cols], zero-filled padding
__global__ __launch_bounds__(BDIM) void cvt_f16_pad_kernel(
    const float* __restrict__ src, _Float16* __restrict__ dst,
    int src_rows, int src_cols, int dst_rows, int dst_cols) {
  size_t total = (size_t)dst_rows * dst_cols;
  size_t i = (size_t)blockIdx.x * BDIM + threadIdx.x;
  size_t stride = (size_t)gridDim.x * BDIM;
  for (; i < total; i += stride) {
    int r = (int)(i / dst_cols);
    int c = (int)(i % dst_cols);
    float v = (r < src_rows && c < src_cols) ? src[(size_t)r * src_cols + c] : 0.f;
    dst[i] = (_Float16)v;
  }
}

// ---------------------------------------------------------------------------
// RMSNorm: one block per row. Writes f16 (GEMM operand) and optional f32 copy.
// ---------------------------------------------------------------------------
__global__ __launch_bounds__(BDIM) void rmsnorm_kernel(const float* __restrict__ x,
                                                       const float* __restrict__ w,
                                                       float* __restrict__ out_f,
                                                       _Float16* __restrict__ out_h,
                                                       int Hd) {
  __shared__ float red[BDIM];
  int row = blockIdx.x;
  const float* xr = x + (size_t)row * Hd;
  float ss = 0.f;
  for (int i = threadIdx.x; i < Hd; i += BDIM) { float v = xr[i]; ss += v * v; }
  red[threadIdx.x] = ss;
  __syncthreads();
  for (int s = BDIM / 2; s > 0; s >>= 1) {
    if (threadIdx.x < (unsigned)s) red[threadIdx.x] += red[threadIdx.x + s];
    __syncthreads();
  }
  float inv = rsqrtf(red[0] / (float)Hd + 1e-6f);
  for (int i = threadIdx.x; i < Hd; i += BDIM) {
    float v = xr[i] * inv * w[i];
    if (out_f) out_f[(size_t)row * Hd + i] = v;
    out_h[(size_t)row * Hd + i] = (_Float16)v;
  }
}

// ---------------------------------------------------------------------------
// WMMA GEMM:  C[M,N] (f32) = A[M,K] (f16 row-major) * W[N,K]^T (f16 row-major)
// Guard-free: requires M % 128 == 0, N % 128 == 0, K % 32 == 0 (callers pad).
// Block tile 128x128, 8 waves in 4x2; each wave a 32x64 micro-tile = 2x4 WMMA
// per 32-wide K step (8 WMMA : 12 ds_load_b128). LDS tiles are double-buffered
// and staged by the Tensor Data Mover: wave 0 issues descriptors for tile k+1
// into the spare buffer at the top of iteration k, the block computes on the
// current buffer, then wave 0 waits TENSORcnt and the block barriers. The TDM
// transfer overlaps the full WMMA phase. TDM pad fields produce 96 B LDS rows
// so v16h fragment loads stay 32-byte aligned.
// ---------------------------------------------------------------------------
#define TM 128
#define TN 128
#define TK 32
#define LDT 48   // halfs per LDS row (32 data + 16 pad)

__global__ __launch_bounds__(BDIM) void wmma_gemm_f16_kernel(
    const _Float16* __restrict__ A, const _Float16* __restrict__ W,
    float* __restrict__ C, int M, int N, int K) {
  __shared__ __align__(128) _Float16 As[2][TM * LDT];
  __shared__ __align__(128) _Float16 Ws[2][TN * LDT];

  const int tid  = threadIdx.x;
  const int lane = tid & 31;
  const int wave = tid >> 5;
  const int bm = blockIdx.y * TM;
  const int bn = blockIdx.x * TN;
  const int wr = (wave & 3) * 32;   // wave row offset inside block tile
  const int wc = (wave >> 2) * 64;  // wave col offset inside block tile

  unsigned lds_a[2], lds_w[2];
  lds_a[0] = (unsigned)(unsigned long long)(void*)&As[0][0];
  lds_a[1] = (unsigned)(unsigned long long)(void*)&As[1][0];
  lds_w[0] = (unsigned)(unsigned long long)(void*)&Ws[0][0];
  lds_w[1] = (unsigned)(unsigned long long)(void*)&Ws[1][0];

  const _Float16* Ab = A + (size_t)bm * K;
  const _Float16* Wb = W + (size_t)bn * K;

  v8f acc[2][4] = {};

  // ---- prologue: stage tile 0 into buffer 0 ------------------------------
  if (wave == 0) {
    tdm_load_tile_f16(Ab, lds_a[0], TK, TM, K, M, K);
    tdm_load_tile_f16(Wb, lds_w[0], TK, TN, K, N, K);
    __builtin_amdgcn_s_wait_tensorcnt(0);
  }
  __syncthreads();

  int cur = 0;
  for (int k0 = 0; k0 < K; k0 += TK) {
    // ---- kick off TDM for tile k+1 into the spare buffer -----------------
    // (safe: everyone finished reading it at the previous barrier)
    if (wave == 0 && k0 + TK < K) {
      tdm_load_tile_f16(Ab + k0 + TK, lds_a[cur ^ 1], TK, TM, K, M, K);
      tdm_load_tile_f16(Wb + k0 + TK, lds_w[cur ^ 1], TK, TN, K, N, K);
    }

    // ---- fragments + WMMA on current buffer ------------------------------
    const int fr = lane & 15;
    const int fk = (lane >> 4) * 16;  // 0 or 16 halfs -> 0 or 32 bytes
    v16h a[2], b[4];
#pragma unroll
    for (int mi = 0; mi < 2; ++mi)
      a[mi] = *(const v16h*)&As[cur][(wr + mi * 16 + fr) * LDT + fk];
#pragma unroll
    for (int ni = 0; ni < 4; ++ni)
      b[ni] = *(const v16h*)&Ws[cur][(wc + ni * 16 + fr) * LDT + fk];

#pragma unroll
    for (int mi = 0; mi < 2; ++mi)
#pragma unroll
      for (int ni = 0; ni < 4; ++ni)
        acc[mi][ni] = __builtin_amdgcn_wmma_f32_16x16x32_f16(
            false, a[mi], false, b[ni], (short)0, acc[mi][ni], false, false);

    // ---- publish next buffer ---------------------------------------------
    if (wave == 0) __builtin_amdgcn_s_wait_tensorcnt(0);
    __syncthreads();
    cur ^= 1;
  }

  // ---- store (16x16 f32 C layout: lanes 0-15 -> M=i, lanes 16-31 -> M=8+i)
  const int cn = lane & 15;
  const int mo = (lane >> 4) * 8;
#pragma unroll
  for (int mi = 0; mi < 2; ++mi) {
#pragma unroll
    for (int ni = 0; ni < 4; ++ni) {
      float* Cp = C + (size_t)(bm + wr + mi * 16 + mo) * N + (bn + wc + ni * 16 + cn);
#pragma unroll
      for (int i = 0; i < 8; ++i) {
        *Cp = acc[mi][ni][i];
        Cp += N;
      }
    }
  }
}

// ---------------------------------------------------------------------------
// Depthwise causal conv (K=4, left pad 3) + bias + SiLU -> f16
// ---------------------------------------------------------------------------
__global__ __launch_bounds__(BDIM) void conv_silu_kernel(
    const float* __restrict__ xn, const float* __restrict__ cw,
    const float* __restrict__ cb, _Float16* __restrict__ out_h) {
  size_t total = (size_t)MS_ * H_;
  size_t idx = (size_t)blockIdx.x * BDIM + threadIdx.x;
  size_t stride = (size_t)gridDim.x * BDIM;
  for (; idx < total; idx += stride) {
    int c = (int)(idx % H_);
    size_t bs = idx / H_;
    int s = (int)(bs % S_);
    size_t b = bs / S_;
    float acc = cb[c];
#pragma unroll
    for (int k = 0; k < K_; ++k) {
      int sp = s - (K_ - 1) + k;
      if (sp >= 0) acc += xn[((size_t)b * S_ + sp) * H_ + c] * cw[c * K_ + k];
    }
    float y = acc / (1.f + __expf(-acc));  // SiLU
    out_h[idx] = (_Float16)y;
  }
}

// ---------------------------------------------------------------------------
// Linear-attention scan. One block per (b, nh). Thread t owns e = t%64 and a
// 16-deep slab of d = [(t/64)*16, +16). State kept in registers across S.
// ---------------------------------------------------------------------------
__global__ __launch_bounds__(BDIM) void scan_kernel(
    const float* __restrict__ qkv, const float* __restrict__ decay_p,
    _Float16* __restrict__ o_h, float* __restrict__ state_out) {
  const int bh = blockIdx.x;
  const int b = bh / NH_;
  const int nh = bh % NH_;
  const int t = threadIdx.x;
  const int e = t & 63;
  const int d0 = (t >> 6) * 16;

  __shared__ float sq[HD_], sk[HD_], sv[HD_];
  __shared__ float pr[4][HD_];

  const float dec = 1.f / (1.f + __expf(-decay_p[nh]));
  float st[16];
#pragma unroll
  for (int j = 0; j < 16; ++j) st[j] = 0.f;

  for (int s = 0; s < S_; ++s) {
    const size_t base = ((size_t)b * S_ + s) * H3_ + (size_t)nh * HD_;
    if (t < HD_) {
      float qr = qkv[base + t];
      float kr = qkv[base + H_ + t];
      sq[t] = qr > 0.f ? qr + 1.f : __expf(qr);   // elu(x)+1
      sk[t] = kr > 0.f ? kr + 1.f : __expf(kr);
      sv[t] = qkv[base + 2 * H_ + t];
    }
    __syncthreads();
    const float ve = sv[e];
    float po = 0.f;
#pragma unroll
    for (int j = 0; j < 16; ++j) {
      st[j] = dec * st[j] + sk[d0 + j] * ve;      // state = decay*state + k v^T
      po += sq[d0 + j] * st[j];                   // partial o[e]
    }
    pr[t >> 6][e] = po;
    __syncthreads();
    if (t < HD_) {
      float o = pr[0][t] + pr[1][t] + pr[2][t] + pr[3][t];
      float qs = 0.f, ks = 0.f;
      for (int i = 0; i < HD_; ++i) { qs += sq[i]; ks += sk[i]; }
      float nrm = fmaxf(qs * ks, 1e-6f);          // sum(q)*sum(k)
      o_h[((size_t)b * S_ + s) * H_ + (size_t)nh * HD_ + t] = (_Float16)(o / nrm);
    }
    __syncthreads();
  }
  // final state [B, NH, HD, HD]
#pragma unroll
  for (int j = 0; j < 16; ++j)
    state_out[(((size_t)b * NH_ + nh) * HD_ + (d0 + j)) * HD_ + e] = st[j];
}

// ---------------------------------------------------------------------------
// concat(local, global) -> f16 [MS, 2H]
// ---------------------------------------------------------------------------
__global__ __launch_bounds__(BDIM) void concat_kernel(
    const float* __restrict__ loc, const float* __restrict__ glb,
    _Float16* __restrict__ comb) {
  size_t total = (size_t)MS_ * H2_;
  size_t idx = (size_t)blockIdx.x * BDIM + threadIdx.x;
  size_t stride = (size_t)gridDim.x * BDIM;
  for (; idx < total; idx += stride) {
    size_t row = idx / H2_;
    int c = (int)(idx % H2_);
    float v = (c < H_) ? loc[row * H_ + c] : glb[row * H_ + (c - H_)];
    comb[idx] = (_Float16)v;
  }
}

// x1 = x + g*local + (1-g)*global, g = sigmoid(gpre)
__global__ __launch_bounds__(BDIM) void combine_kernel(
    const float* __restrict__ x, const float* __restrict__ loc,
    const float* __restrict__ glb, const float* __restrict__ gpre,
    float* __restrict__ x1) {
  size_t total = (size_t)MS_ * H_;
  size_t idx = (size_t)blockIdx.x * BDIM + threadIdx.x;
  size_t stride = (size_t)gridDim.x * BDIM;
  for (; idx < total; idx += stride) {
    float g = 1.f / (1.f + __expf(-gpre[idx]));
    x1[idx] = x[idx] + g * loc[idx] + (1.f - g) * glb[idx];
  }
}

// prod = f16( silu(gp) * up ) over padded [MS, IP]. Pad cols are zero because
// the padded weight rows are zero (silu(0)*0 == 0), so no branch needed.
__global__ __launch_bounds__(BDIM) void ffn_act_kernel(
    const float* __restrict__ gp, const float* __restrict__ up,
    _Float16* __restrict__ prod) {
  size_t total = (size_t)MS_ * IP_;
  size_t idx = (size_t)blockIdx.x * BDIM + threadIdx.x;
  size_t stride = (size_t)gridDim.x * BDIM;
  for (; idx < total; idx += stride) {
    float a = gp[idx];
    float s = a / (1.f + __expf(-a));
    prod[idx] = (_Float16)(s * up[idx]);
  }
}

// out = x1 + ffn
__global__ __launch_bounds__(BDIM) void final_add_kernel(
    const float* __restrict__ x1, const float* __restrict__ ffn,
    float* __restrict__ out) {
  size_t total = (size_t)MS_ * H_;
  size_t idx = (size_t)blockIdx.x * BDIM + threadIdx.x;
  size_t stride = (size_t)gridDim.x * BDIM;
  for (; idx < total; idx += stride) out[idx] = x1[idx] + ffn[idx];
}

// ---------------------------------------------------------------------------
// Host-side orchestration
// ---------------------------------------------------------------------------
static inline dim3 gemm_grid(int M, int N) {
  return dim3(N / TN, M / TM);
}

extern "C" void kernel_launch(void* const* d_in, const int* in_sizes, int n_in,
                              void* d_out, int out_size, void* d_ws, size_t ws_size,
                              hipStream_t stream) {
  (void)in_sizes; (void)n_in; (void)out_size; (void)ws_size;

  const float* x        = (const float*)d_in[0];
  const float* qkv_w    = (const float*)d_in[1];
  const float* out_w    = (const float*)d_in[2];
  const float* decay_p  = (const float*)d_in[3];
  const float* conv_w   = (const float*)d_in[4];
  const float* conv_b   = (const float*)d_in[5];
  const float* pw_w     = (const float*)d_in[6];
  const float* gate_w   = (const float*)d_in[7];
  const float* gp_w     = (const float*)d_in[8];
  const float* up_w     = (const float*)d_in[9];
  const float* dp_w     = (const float*)d_in[10];
  const float* norm1_w  = (const float*)d_in[11];
  const float* norm2_w  = (const float*)d_in[12];

  float* out_main  = (float*)d_out;                       // [MS, H]
  float* out_state = (float*)d_out + (size_t)MS_ * H_;    // [B, NH, HD, HD]

  // ---- bump allocator over workspace -------------------------------------
  char* p = (char*)d_ws;
  auto alloc = [&](size_t bytes) -> void* {
    void* r = (void*)p;
    p += (bytes + 255) & ~(size_t)255;
    return r;
  };

  _Float16* qkvw_h  = (_Float16*)alloc((size_t)H3_ * H_ * 2);
  _Float16* outw_h  = (_Float16*)alloc((size_t)H_ * H_ * 2);
  _Float16* pww_h   = (_Float16*)alloc((size_t)H_ * H_ * 2);
  _Float16* gatew_h = (_Float16*)alloc((size_t)H_ * H2_ * 2);
  _Float16* gpw_h   = (_Float16*)alloc((size_t)IP_ * H_ * 2);   // pad rows I..IP
  _Float16* upw_h   = (_Float16*)alloc((size_t)IP_ * H_ * 2);   // pad rows I..IP
  _Float16* dpw_h   = (_Float16*)alloc((size_t)H_ * IP_ * 2);   // pad cols I..IP

  _Float16* xn_h    = (_Float16*)alloc((size_t)MS_ * H_ * 2);
  float*    xn_f    = (float*)   alloc((size_t)MS_ * H_ * 4);
  float*    qkv_f   = (float*)   alloc((size_t)MS_ * H3_ * 4);
  _Float16* convs_h = (_Float16*)alloc((size_t)MS_ * H_ * 2);
  _Float16* o_h     = (_Float16*)alloc((size_t)MS_ * H_ * 2);
  float*    local_f = (float*)   alloc((size_t)MS_ * H_ * 4);
  float*    glob_f  = (float*)   alloc((size_t)MS_ * H_ * 4);
  _Float16* comb_h  = (_Float16*)alloc((size_t)MS_ * H2_ * 2);
  float*    gpre_f  = (float*)   alloc((size_t)MS_ * H_ * 4);
  float*    x1_f    = (float*)   alloc((size_t)MS_ * H_ * 4);
  _Float16* xn2_h   = (_Float16*)alloc((size_t)MS_ * H_ * 2);
  float*    up_f    = (float*)   alloc((size_t)MS_ * IP_ * 4);
  _Float16* prod_h  = (_Float16*)alloc((size_t)MS_ * IP_ * 2);
  float*    ffn_f   = (float*)   alloc((size_t)MS_ * H_ * 4);
  float*    gp_f    = qkv_f;  // reuse: qkv consumed by scan before FFN starts
                              // (MS*IP*4 <= MS*3H*4 since IP < 3H)

  const int EW = 2048;  // elementwise grid blocks

  // 1) RMSNorm #1 (f16 + f32 copies)
  rmsnorm_kernel<<<MS_, BDIM, 0, stream>>>(x, norm1_w, xn_f, xn_h, H_);

  // 2) weights -> f16 (FFN matrices zero-padded to IP)
  cvt_f16_kernel<<<EW, BDIM, 0, stream>>>(qkv_w, qkvw_h, (size_t)H3_ * H_);
  cvt_f16_kernel<<<EW, BDIM, 0, stream>>>(out_w, outw_h, (size_t)H_ * H_);
  cvt_f16_kernel<<<EW, BDIM, 0, stream>>>(pw_w,  pww_h,  (size_t)H_ * H_);
  cvt_f16_kernel<<<EW, BDIM, 0, stream>>>(gate_w, gatew_h, (size_t)H_ * H2_);
  cvt_f16_pad_kernel<<<EW, BDIM, 0, stream>>>(gp_w, gpw_h, I_, H_, IP_, H_);
  cvt_f16_pad_kernel<<<EW, BDIM, 0, stream>>>(up_w, upw_h, I_, H_, IP_, H_);
  cvt_f16_pad_kernel<<<EW, BDIM, 0, stream>>>(dp_w, dpw_h, H_, I_, H_, IP_);

  // 3) qkv = xn @ qkv_w^T   [MS, 3H]
  wmma_gemm_f16_kernel<<<gemm_grid(MS_, H3_), BDIM, 0, stream>>>(
      xn_h, qkvw_h, qkv_f, MS_, H3_, H_);

  // 4) local path: conv+silu then pw GEMM
  conv_silu_kernel<<<EW, BDIM, 0, stream>>>(xn_f, conv_w, conv_b, convs_h);
  wmma_gemm_f16_kernel<<<gemm_grid(MS_, H_), BDIM, 0, stream>>>(
      convs_h, pww_h, local_f, MS_, H_, H_);

  // 5) attention scan (o + final state)
  scan_kernel<<<B_ * NH_, BDIM, 0, stream>>>(qkv_f, decay_p, o_h, out_state);

  // 6) global_out = o @ out_w^T
  wmma_gemm_f16_kernel<<<gemm_grid(MS_, H_), BDIM, 0, stream>>>(
      o_h, outw_h, glob_f, MS_, H_, H_);

  // 7) gate
  concat_kernel<<<EW, BDIM, 0, stream>>>(local_f, glob_f, comb_h);
  wmma_gemm_f16_kernel<<<gemm_grid(MS_, H_), BDIM, 0, stream>>>(
      comb_h, gatew_h, gpre_f, MS_, H_, H2_);
  combine_kernel<<<EW, BDIM, 0, stream>>>(x, local_f, glob_f, gpre_f, x1_f);

  // 8) RMSNorm #2 + FFN (padded to IP; pad region stays zero end-to-end)
  rmsnorm_kernel<<<MS_, BDIM, 0, stream>>>(x1_f, norm2_w, (float*)nullptr, xn2_h, H_);
  wmma_gemm_f16_kernel<<<gemm_grid(MS_, IP_), BDIM, 0, stream>>>(
      xn2_h, gpw_h, gp_f, MS_, IP_, H_);
  wmma_gemm_f16_kernel<<<gemm_grid(MS_, IP_), BDIM, 0, stream>>>(
      xn2_h, upw_h, up_f, MS_, IP_, H_);
  ffn_act_kernel<<<EW, BDIM, 0, stream>>>(gp_f, up_f, prod_h);
  wmma_gemm_f16_kernel<<<gemm_grid(MS_, H_), BDIM, 0, stream>>>(
      prod_h, dpw_h, ffn_f, MS_, H_, IP_);

  // 9) residual
  final_add_kernel<<<EW, BDIM, 0, stream>>>(x1_f, ffn_f, out_main);
}
